// parallel_encoder_32401233281582
// MI455X (gfx1250) — matrix operand order repaired and checked by reading, not compiled, not run on hardware
//
#include <hip/hip_runtime.h>

// ---------------------------------------------------------------------------
// CDNA5 (gfx1250) bf16 WMMA GEMM + GCN scatter pipeline
// ---------------------------------------------------------------------------

typedef __attribute__((ext_vector_type(16))) __bf16 v16bf;
typedef __attribute__((ext_vector_type(8)))  float  v8f;

union Frag16 { v16bf v; uint4 q[2]; };

#if defined(__has_builtin)
#if __has_builtin(__builtin_amdgcn_sched_barrier)
#define SCHED_FENCE() __builtin_amdgcn_sched_barrier(0)
#endif
#endif
#ifndef SCHED_FENCE
#define SCHED_FENCE()
#endif

__device__ __forceinline__ unsigned short f2bf(float f) {
  union { float f; unsigned int u; } c; c.f = f;
  unsigned int r = c.u + 0x7FFFu + ((c.u >> 16) & 1u);  // round-to-nearest-even
  return (unsigned short)(r >> 16);
}

#define APITCH 40  // halves per LDS row (32 data + 8 pad) -> 80B rows, 16B aligned chunks

// ---------------------------------------------------------------------------
// GEMM: C[M,N] = act(A[M,K](f32) * Bt[N,Kp](bf16,row=n) + bias)
// Block = 256 threads = 8 waves; wave (wm,wn) owns rows [wm*16,+16) and
// WNT 16-wide n-tiles. Per k-step: issue all fragment ds_loads, one dscnt
// wait, then WNT back-to-back v_wmma_f32_16x16x32_bf16 (sched_barrier-fenced
// so the scheduler cannot re-serialize load->wait->wmma per tile).
// ---------------------------------------------------------------------------
template<int WM, int WN, int WNT>
__global__ __launch_bounds__(256) void gemm_bf16_wmma(
    const float* __restrict__ A, const unsigned short* __restrict__ Bt,
    const float* __restrict__ bias, float* __restrict__ C,
    int M, int N, int K, int Kp, int do_relu)
{
  constexpr int BM = 16 * WM;
  constexpr int BN = 16 * WNT * WN;
  __shared__ unsigned short sA[BM * APITCH];
  __shared__ unsigned short sB[BN * APITCH];

  const int tid  = threadIdx.x;
  const int wave = tid >> 5;
  const int lane = tid & 31;
  const int wm   = wave / WN;
  const int wn   = wave % WN;
  const int grp  = lane >> 4;   // half-wave group (ISA fragment layout)
  const int lr   = lane & 15;
  const int m0   = blockIdx.x * BM;
  const int n0   = blockIdx.y * BN;

  // Interior-block fast path predicates (uniform across block)
  const bool mFull  = (m0 + BM) <= M;
  const bool kVec   = (K & 3) == 0;

  v8f acc[WNT];
  v8f zero = {0.f, 0.f, 0.f, 0.f, 0.f, 0.f, 0.f, 0.f};
#pragma unroll
  for (int t = 0; t < WNT; ++t) acc[t] = zero;

  for (int k0 = 0; k0 < K; k0 += 32) {
    __syncthreads();
    const bool kFull = (k0 + 32) <= K;
    // ---- stage A tile: f32 -> bf16 into LDS [BM][32] ----
    if (mFull && kFull && kVec) {
      for (int c = tid; c < BM * 4; c += 256) {
        const int row = c >> 2;
        const int ck  = (c & 3) * 8;
        const float4* src = (const float4*)(A + (long long)(m0 + row) * K + k0 + ck);
        float4 f0 = src[0];
        float4 f1 = src[1];
        __builtin_prefetch(src + 8, 0, 1);  // global_prefetch_b8: next k-block
        uint4 p;
        p.x = (unsigned)f2bf(f0.x) | ((unsigned)f2bf(f0.y) << 16);
        p.y = (unsigned)f2bf(f0.z) | ((unsigned)f2bf(f0.w) << 16);
        p.z = (unsigned)f2bf(f1.x) | ((unsigned)f2bf(f1.y) << 16);
        p.w = (unsigned)f2bf(f1.z) | ((unsigned)f2bf(f1.w) << 16);
        *(uint4*)&sA[row * APITCH + ck] = p;
      }
    } else {
      for (int c = tid; c < BM * 4; c += 256) {
        const int row = c >> 2;
        const int ck  = (c & 3) * 8;
        const int m   = m0 + row;
        unsigned short h[8];
#pragma unroll
        for (int j = 0; j < 8; ++j) {
          int k = k0 + ck + j;
          h[j] = (m < M && k < K) ? f2bf(A[(long long)m * K + k]) : (unsigned short)0;
        }
        uint4 p;
        p.x = (unsigned)h[0] | ((unsigned)h[1] << 16);
        p.y = (unsigned)h[2] | ((unsigned)h[3] << 16);
        p.z = (unsigned)h[4] | ((unsigned)h[5] << 16);
        p.w = (unsigned)h[6] | ((unsigned)h[7] << 16);
        *(uint4*)&sA[row * APITCH + ck] = p;
      }
    }
    // ---- stage B tile: bf16 (pre-transposed [N][Kp], zero-padded K) ----
    for (int c = tid; c < BN * 4; c += 256) {
      const int row = c >> 2;
      const int ck  = (c & 3) * 8;
      const int nn  = n0 + row;
      uint4 p;
      if (nn < N) p = *(const uint4*)(Bt + (long long)nn * Kp + k0 + ck);
      else { p.x = 0u; p.y = 0u; p.z = 0u; p.w = 0u; }
      *(uint4*)&sB[row * APITCH + ck] = p;
    }
    __syncthreads();

    // ---- A fragment: lane lr = row m; halves k = {grp*8..+7, 16+grp*8..+7} ----
    Frag16 fa;
    {
      const unsigned short* ap = &sA[(wm * 16 + lr) * APITCH + grp * 8];
      fa.q[0] = *(const uint4*)(ap);
      fa.q[1] = *(const uint4*)(ap + 16);
    }
    // ---- preload ALL B fragments (kept live across the WMMA chain) ----
    Frag16 fb[WNT];
#pragma unroll
    for (int t = 0; t < WNT; ++t) {
      const unsigned short* bp = &sB[((wn * WNT + t) * 16 + lr) * APITCH + grp * 8];
      fb[t].q[0] = *(const uint4*)(bp);
      fb[t].q[1] = *(const uint4*)(bp + 16);
    }
    SCHED_FENCE();  // all ds_loads above issue first -> single s_wait_dscnt
#pragma unroll
    for (int t = 0; t < WNT; ++t) {
      acc[t] = __builtin_amdgcn_wmma_f32_16x16x32_bf16(
          false, fa.v, false, fb[t].v, (short)0, acc[t], false, false);
    }
    SCHED_FENCE();  // keep the WMMA burst contiguous
  }

  // ---- store: C/D layout -> VGPR r holds row m0+wm*16+grp*8+r, col n0+...+lr ----
#pragma unroll
  for (int t = 0; t < WNT; ++t) {
    const int nn = n0 + (wn * WNT + t) * 16 + lr;
    const float bv = (bias != nullptr && nn < N) ? bias[nn] : 0.f;
#pragma unroll
    for (int r = 0; r < 8; ++r) {
      const int mm = m0 + wm * 16 + grp * 8 + r;
      if (mm < M && nn < N) {
        float v = acc[t][r] + bv;
        if (do_relu) v = fmaxf(v, 0.f);
        C[(long long)mm * N + nn] = v;
      }
    }
  }
}

// ---------------------------------------------------------------------------
// Support kernels
// ---------------------------------------------------------------------------
__global__ void conv_wt_bf16(const float* __restrict__ W, unsigned short* __restrict__ Wt,
                             int K, int N, int Kp)  // W:[K][N] -> Wt:[N][Kp] bf16, zero-padded
{
  long long total = (long long)N * Kp;
  long long t = (long long)blockIdx.x * blockDim.x + threadIdx.x;
  if (t >= total) return;
  int nn = (int)(t / Kp);
  int k  = (int)(t % Kp);
  Wt[t] = (k < K) ? f2bf(W[(long long)k * N + nn]) : (unsigned short)0;
}

__global__ void fill_f32(float* __restrict__ p, float v, long long total) {
  long long t = (long long)blockIdx.x * blockDim.x + threadIdx.x;
  if (t < total) p[t] = v;
}

__global__ void deg_count(const int* __restrict__ dst, float* __restrict__ deg, int E) {
  int t = blockIdx.x * blockDim.x + threadIdx.x;
  if (t < E) atomicAdd(&deg[dst[t]], 1.0f);
}

__global__ void rsqrt_inplace(float* __restrict__ p, int n) {
  int t = blockIdx.x * blockDim.x + threadIdx.x;
  if (t < n) p[t] = rsqrtf(fmaxf(p[t], 1e-12f));
}

// coef = cadd + cmul * (*cptr); one wave per edge, lanes stride features.
__global__ void scatter_edges(const int* __restrict__ src, const int* __restrict__ dst,
                              const float* __restrict__ h, const float* __restrict__ dinv,
                              float* __restrict__ out, int E, int F,
                              const float* __restrict__ cptr, float cmul, float cadd)
{
  const int e = blockIdx.x * (blockDim.x >> 5) + (threadIdx.x >> 5);
  if (e >= E) return;
  const float w    = cptr ? *cptr : 0.f;
  const float coef = cadd + cmul * w;
  const int s = src[e];
  const int d = dst[e];
  const float nrm = coef * dinv[s] * dinv[d];
  const int lane = threadIdx.x & 31;
  const float* hp = h + (long long)s * F;
  float* op = out + (long long)d * F;
  for (int f = lane; f < F; f += 32) atomicAdd(op + f, nrm * hp[f]);
}

__global__ void self_loop_add(const float* __restrict__ h, const float* __restrict__ dinv,
                              float* __restrict__ out, long long n, int F,
                              const float* __restrict__ cptr, float cmul, float cadd)
{
  long long total = n * F;
  long long t = (long long)blockIdx.x * blockDim.x + threadIdx.x;
  if (t >= total) return;
  const float w    = cptr ? *cptr : 0.f;
  const float coef = cadd + cmul * w;
  const int i = (int)(t / F);
  const float dv = dinv[i];
  out[t] += coef * dv * dv * h[t];
}

__global__ void bias_relu_inplace(float* __restrict__ x, const float* __restrict__ b,
                                  long long n, int F) {
  long long total = n * F;
  long long t = (long long)blockIdx.x * blockDim.x + threadIdx.x;
  if (t >= total) return;
  x[t] = fmaxf(x[t] + b[(int)(t % F)], 0.f);
}

__global__ void blend(const float* __restrict__ a, const float* __restrict__ b,
                      float* __restrict__ o, const float* __restrict__ wp, long long total) {
  long long t = (long long)blockIdx.x * blockDim.x + threadIdx.x;
  if (t >= total) return;
  const float w = *wp;
  o[t] = w * a[t] + (1.f - w) * b[t];
}

__global__ void init_out_bias(float* __restrict__ out, const float* __restrict__ b1,
                              const float* __restrict__ b2, const float* __restrict__ wp,
                              long long n, int F) {
  long long total = n * F;
  long long t = (long long)blockIdx.x * blockDim.x + threadIdx.x;
  if (t >= total) return;
  const float w = *wp;
  const int f = (int)(t % F);
  out[t] = w * b1[f] + (1.f - w) * b2[f];
}

// ---------------------------------------------------------------------------
// Host orchestration
// ---------------------------------------------------------------------------
extern "C" void kernel_launch(void* const* d_in, const int* in_sizes, int n_in,
                              void* d_out, int out_size, void* d_ws, size_t ws_size,
                              hipStream_t stream)
{
  const float* gene = (const float*)d_in[0];
  const float* img  = (const float*)d_in[1];
  const int*   ge_ei = (const int*)d_in[2];
  const int*   sp_ei = (const int*)d_in[3];
  const float* Wfc1 = (const float*)d_in[4];  const float* bfc1 = (const float*)d_in[5];
  const float* Wfc2 = (const float*)d_in[6];  const float* bfc2 = (const float*)d_in[7];
  const float* Wg11 = (const float*)d_in[8];  const float* bg11 = (const float*)d_in[9];
  const float* Wg12 = (const float*)d_in[10]; const float* bg12 = (const float*)d_in[11];
  const float* Wg21 = (const float*)d_in[12]; const float* bg21 = (const float*)d_in[13];
  const float* Wg22 = (const float*)d_in[14]; const float* bg22 = (const float*)d_in[15];
  const float* w1p  = (const float*)d_in[16];
  const float* w2p  = (const float*)d_in[17];

  const int n   = in_sizes[0] / 3000;   // nodes
  const int Ege = in_sizes[2] / 2;
  const int Esp = in_sizes[3] / 2;
  const int Kp1 = 3008;                 // 3000 rounded up to 32

  // workspace carve-out (256B aligned)
  char* ws = (char*)d_ws;
  size_t off = 0;
  auto alloc = [&](size_t bytes) -> void* {
    void* p = ws + off;
    off = (off + bytes + 255) & ~(size_t)255;
    return p;
  };
  float* x1    = (float*)alloc((size_t)n * 512 * 4);
  float* x2    = (float*)alloc((size_t)n * 512 * 4);
  float* h1pre = (float*)alloc((size_t)n * 256 * 4);
  float* h2pre = (float*)alloc((size_t)n * 256 * 4);
  unsigned short* Wfc1t = (unsigned short*)alloc((size_t)512 * Kp1 * 2);
  unsigned short* Wfc2t = (unsigned short*)alloc((size_t)512 * 1024 * 2);
  unsigned short* Wg11t = (unsigned short*)alloc((size_t)256 * 512 * 2);
  unsigned short* Wg12t = (unsigned short*)alloc((size_t)256 * 512 * 2);
  unsigned short* Wg21t = (unsigned short*)alloc((size_t)64 * 256 * 2);
  unsigned short* Wg22t = (unsigned short*)alloc((size_t)64 * 256 * 2);
  float* dsp = (float*)alloc((size_t)n * 4);
  float* dge = (float*)alloc((size_t)n * 4);
  // aliases (live ranges do not overlap; stream-ordered)
  float* agg1  = x1;                         // after x1 dead
  float* agg2  = x1 + (size_t)n * 256;
  float* xc    = x2;                         // after x2 dead
  float* o1pre = x2 + (size_t)n * 256;
  float* o2pre = x2 + (size_t)n * 320;
  float* out   = (float*)d_out;

  auto cdiv = [](long long a, long long b) { return (int)((a + b - 1) / b); };

  // 1) weights: f32 [K][N] -> bf16 [N][Kp]
  conv_wt_bf16<<<cdiv((long long)512 * Kp1, 256), 256, 0, stream>>>(Wfc1, Wfc1t, 3000, 512, Kp1);
  conv_wt_bf16<<<cdiv((long long)512 * 1024, 256), 256, 0, stream>>>(Wfc2, Wfc2t, 1024, 512, 1024);
  conv_wt_bf16<<<cdiv((long long)256 * 512, 256), 256, 0, stream>>>(Wg11, Wg11t, 512, 256, 512);
  conv_wt_bf16<<<cdiv((long long)256 * 512, 256), 256, 0, stream>>>(Wg12, Wg12t, 512, 256, 512);
  conv_wt_bf16<<<cdiv((long long)64 * 256, 256), 256, 0, stream>>>(Wg21, Wg21t, 256, 64, 256);
  conv_wt_bf16<<<cdiv((long long)64 * 256, 256), 256, 0, stream>>>(Wg22, Wg22t, 256, 64, 256);

  // 2) degrees (self loop => init 1) -> dinv
  fill_f32<<<cdiv(n, 256), 256, 0, stream>>>(dsp, 1.0f, n);
  fill_f32<<<cdiv(n, 256), 256, 0, stream>>>(dge, 1.0f, n);
  deg_count<<<cdiv(Esp, 256), 256, 0, stream>>>(sp_ei + Esp, dsp, Esp);
  deg_count<<<cdiv(Ege, 256), 256, 0, stream>>>(ge_ei + Ege, dge, Ege);
  rsqrt_inplace<<<cdiv(n, 256), 256, 0, stream>>>(dsp, n);
  rsqrt_inplace<<<cdiv(n, 256), 256, 0, stream>>>(dge, n);

  // 3) encoders: x1 = relu(gene@Wfc1+b), x2 = relu(img@Wfc2+b)   [N tile = 512]
  gemm_bf16_wmma<2, 4, 8><<<dim3(cdiv(n, 32), 1), 256, 0, stream>>>(
      gene, Wfc1t, bfc1, x1, n, 512, 3000, Kp1, 1);
  gemm_bf16_wmma<2, 4, 8><<<dim3(cdiv(n, 32), 1), 256, 0, stream>>>(
      img, Wfc2t, bfc2, x2, n, 512, 1024, 1024, 1);

  // 4) GCN layer-1 feature transforms (bias added post-aggregation)  [N tile = 256]
  gemm_bf16_wmma<4, 2, 8><<<dim3(cdiv(n, 64), 1), 256, 0, stream>>>(
      x1, Wg11t, nullptr, h1pre, n, 256, 512, 512, 0);
  gemm_bf16_wmma<4, 2, 8><<<dim3(cdiv(n, 64), 1), 256, 0, stream>>>(
      x2, Wg12t, nullptr, h2pre, n, 256, 512, 512, 0);

  // 5) aggregate layer-1 (h1: spatial graph, h2: gene graph), then relu(.+b)
  fill_f32<<<cdiv((long long)n * 256, 256), 256, 0, stream>>>(agg1, 0.f, (long long)n * 256);
  fill_f32<<<cdiv((long long)n * 256, 256), 256, 0, stream>>>(agg2, 0.f, (long long)n * 256);
  scatter_edges<<<cdiv(Esp, 8), 256, 0, stream>>>(sp_ei, sp_ei + Esp, h1pre, dsp, agg1,
                                                  Esp, 256, nullptr, 0.f, 1.f);
  scatter_edges<<<cdiv(Ege, 8), 256, 0, stream>>>(ge_ei, ge_ei + Ege, h2pre, dge, agg2,
                                                  Ege, 256, nullptr, 0.f, 1.f);
  self_loop_add<<<cdiv((long long)n * 256, 256), 256, 0, stream>>>(h1pre, dsp, agg1, n, 256,
                                                                   nullptr, 0.f, 1.f);
  self_loop_add<<<cdiv((long long)n * 256, 256), 256, 0, stream>>>(h2pre, dge, agg2, n, 256,
                                                                   nullptr, 0.f, 1.f);
  bias_relu_inplace<<<cdiv((long long)n * 256, 256), 256, 0, stream>>>(agg1, bg11, n, 256);
  bias_relu_inplace<<<cdiv((long long)n * 256, 256), 256, 0, stream>>>(agg2, bg12, n, 256);

  // 6) x = w1*h1 + (1-w1)*h2
  blend<<<cdiv((long long)n * 256, 256), 256, 0, stream>>>(agg1, agg2, xc, w1p, (long long)n * 256);

  // 7) GCN layer-2 feature transforms  [N tile = 64]
  gemm_bf16_wmma<8, 1, 4><<<dim3(cdiv(n, 128), 1), 256, 0, stream>>>(
      xc, Wg21t, nullptr, o1pre, n, 64, 256, 256, 0);
  gemm_bf16_wmma<8, 1, 4><<<dim3(cdiv(n, 128), 1), 256, 0, stream>>>(
      xc, Wg22t, nullptr, o2pre, n, 64, 256, 256, 0);

  // 8) out = w2*(agg_sp(o1)+b21) + (1-w2)*(agg_ge(o2)+b22), fused into scatter
  init_out_bias<<<cdiv((long long)n * 64, 256), 256, 0, stream>>>(out, bg21, bg22, w2p, n, 64);
  scatter_edges<<<cdiv(Esp, 8), 256, 0, stream>>>(sp_ei, sp_ei + Esp, o1pre, dsp, out,
                                                  Esp, 64, w2p, 1.f, 0.f);
  self_loop_add<<<cdiv((long long)n * 64, 256), 256, 0, stream>>>(o1pre, dsp, out, n, 64,
                                                                  w2p, 1.f, 0.f);
  scatter_edges<<<cdiv(Ege, 8), 256, 0, stream>>>(ge_ei, ge_ei + Ege, o2pre, dge, out,
                                                  Ege, 64, w2p, -1.f, 1.f);
  self_loop_add<<<cdiv((long long)n * 64, 256), 256, 0, stream>>>(o2pre, dge, out, n, 64,
                                                                  w2p, -1.f, 1.f);
}
